// TimeAwareNodeModel_59313498358413
// MI455X (gfx1250) — matrix operand order repaired and checked by reading, not compiled, not run on hardware
//
#include <hip/hip_runtime.h>
#include <hip/hip_bf16.h>
#include <math.h>

// ---------------- problem constants (from reference) ----------------
#define NN      100000
#define DD      128
#define EDIM    32
#define EFULL   1000000
#define EHALF   500000
#define HH      2
#define HID     128
#define DOUT    64
#define FIN     160          // D + EDIM
#define NHID    256
#define NOUT    128
#define ALPHA   0.2f

typedef __attribute__((ext_vector_type(16))) __bf16 v16bf;
typedef __attribute__((ext_vector_type(8)))  float  v8f;
typedef __attribute__((ext_vector_type(4)))  unsigned int u32x4;
typedef __attribute__((ext_vector_type(8)))  int          i32x8;
typedef __attribute__((ext_vector_type(4)))  int          i32x4;

// ---- WMMA fragment loads as wide vector loads (CDNA5 ISA 7.12.2 layouts) ----
// A (16-bit, 16x32): lane holds row M=lane&15; elements j=0..7 -> K=koff..koff+7,
//                    j=8..15 -> K=koff+16..koff+23, with koff = kt*32 + 8*(lane>>4).
// B (32x16):         lane holds col N=lane&15; same K order as A (koff..koff+7, koff+16..koff+23),
//                    when the weight matrix is stored TRANSPOSED ([col][K]).
union fragU { v16bf v; uint4 q[2]; };
__device__ __forceinline__ v16bf fragA(const __bf16* rowbase, int koff) {
  fragU u;
  u.q[0] = *(const uint4*)(rowbase + koff);
  u.q[1] = *(const uint4*)(rowbase + koff + 16);
  return u.v;
}
__device__ __forceinline__ v16bf fragBT(const __bf16* colbase, int koff) {
  fragU u;
  u.q[0] = *(const uint4*)(colbase + koff);
  u.q[1] = *(const uint4*)(colbase + koff + 16);
  return u.v;
}

__device__ __forceinline__ v8f wmma_bf16(v16bf a, v16bf b, v8f c) {
  return __builtin_amdgcn_wmma_f32_16x16x32_bf16(false, a, false, b, (short)0, c, false, false);
}

// ---------------- Tensor Data Mover: 1-D DMA of 16-bit elements into LDS ----------------
// D# packing per CDNA5 ISA 08_async_tensor §8.3/8.4 (group0 128b, group1 256b).
// This toolchain (clang-23/therock-10.0) declares the 6-arg builtin:
//   (u32x4 g0, i32x8 g1, i32x4 g2, i32x4 g3, i32x8 g4, i32 cpol)
__device__ __forceinline__ unsigned lds_off(const void* p) {
  // generic LDS pointer = {shared aperture, 32-bit LDS byte offset}
  return (unsigned)(unsigned long long)p;
}
__device__ __forceinline__ void tdm_load_2B(unsigned lds_addr, const void* gsrc, unsigned nelem) {
  unsigned long long ga = (unsigned long long)gsrc;
  u32x4 g0;
  g0[0] = 1u;                                    // count=1, user descriptor, load
  g0[1] = lds_addr;                              // lds_addr (bytes)
  g0[2] = (unsigned)(ga & 0xffffffffu);          // global_addr[31:0]
  g0[3] = (unsigned)((ga >> 32) & 0x01ffffffu)   // global_addr[56:32]
          | (2u << 30);                          // type=2 ("image")
  i32x8 g1;
  g1[0] = (int)(1u << 16);                       // workgroup_mask=0, data_size=1 (2 bytes)
  g1[1] = (int)((nelem & 0xffffu) << 16);        // tensor_dim0[15:0]
  g1[2] = (int)(((nelem >> 16) & 0xffffu)        // tensor_dim0[31:16]
          | (1u << 16));                         // tensor_dim1 = 1
  g1[3] = (int)((nelem & 0xffffu) << 16);        // tile_dim0 = nelem (fits 16 bits)
  g1[4] = 0;                                     // tile_dim1 = tile_dim2 = 0 (1-D)
  g1[5] = (int)nelem;                            // tensor_dim0_stride[31:0]
  g1[6] = 0;                                     // stride hi, tensor_dim1_stride lo
  g1[7] = 0;
  i32x4 gz4 = {0, 0, 0, 0};                      // groups 2/3 unused (<=2-D tensor)
  i32x8 gz8 = {0, 0, 0, 0, 0, 0, 0, 0};
  __builtin_amdgcn_tensor_load_to_lds(g0, g1, gz4, gz4, gz8, 0);
}

// ordered-int encoding for float atomic max
__device__ __forceinline__ unsigned fkey(float f) {
  unsigned b = __float_as_uint(f);
  return (b & 0x80000000u) ? ~b : (b | 0x80000000u);
}
__device__ __forceinline__ float fdec(unsigned k) {
  return (k & 0x80000000u) ? __uint_as_float(k ^ 0x80000000u) : __uint_as_float(~k);
}
#define NEGINF_KEY 0x007FFFFFu   // fkey(-inf)

// ---------------- elementwise f32 -> bf16 ----------------
__global__ void cvt_bf16_kernel(const float* __restrict__ in, __bf16* __restrict__ out, int n) {
  int i = blockIdx.x * blockDim.x + threadIdx.x;
  if (i < n) out[i] = (__bf16)in[i];
}

// ---------------- f32 [nmat][R][C] -> bf16 [nmat][C][R] (transposed for B fragments) ----------------
__global__ void cvt_bf16_T_kernel(const float* __restrict__ in, __bf16* __restrict__ out,
                                  int R, int C, int nmat) {
  int i = blockIdx.x * blockDim.x + threadIdx.x;
  if (i >= nmat * R * C) return;
  int m = i / (R * C), rc = i - m * (R * C);
  int r = rc / C, c = rc - r * C;
  out[(size_t)m * R * C + (size_t)c * R + r] = (__bf16)in[i];
}

// ---------------- init softmax state ----------------
__global__ void init_soft_kernel(unsigned* mo, unsigned* mi, float* deno, float* deni) {
  int i = blockIdx.x * blockDim.x + threadIdx.x;
  if (i < NN * HH) { mo[i] = NEGINF_KEY; mi[i] = NEGINF_KEY; deno[i] = 0.f; deni[i] = 0.f; }
}

// ---------------- per-node attention partials: s = aa . x[n] ----------------
__global__ void node_scores_kernel(const float* __restrict__ x,
                                   const float* __restrict__ aao, const float* __restrict__ aai,
                                   float* __restrict__ s1o, float* __restrict__ s2o,
                                   float* __restrict__ s1i, float* __restrict__ s2i) {
  int n = blockIdx.x * blockDim.x + threadIdx.x;
  if (n >= NN) return;
  const float* xr = x + (size_t)n * DD;
  float a0=0,a1=0,a2=0,a3=0,a4=0,a5=0,a6=0,a7=0;
  for (int k = 0; k < DD; ++k) {
    float xv = xr[k];
    a0 = fmaf(aao[k],           xv, a0);  a1 = fmaf(aao[DD + k],       xv, a1);
    a2 = fmaf(aao[256 + k],     xv, a2);  a3 = fmaf(aao[256 + DD + k], xv, a3);
    a4 = fmaf(aai[k],           xv, a4);  a5 = fmaf(aai[DD + k],       xv, a5);
    a6 = fmaf(aai[256 + k],     xv, a6);  a7 = fmaf(aai[256 + DD + k], xv, a7);
  }
  s1o[n*2+0]=a0; s2o[n*2+0]=a1; s1o[n*2+1]=a2; s2o[n*2+1]=a3;
  s1i[n*2+0]=a4; s2i[n*2+0]=a5; s1i[n*2+1]=a6; s2i[n*2+1]=a7;
}

// ---------------- logits + segment max ----------------
__global__ void edge_logits_kernel(const int* __restrict__ ei,
                                   const float* __restrict__ s1o, const float* __restrict__ s2o,
                                   const float* __restrict__ s1i, const float* __restrict__ s2i,
                                   float* __restrict__ att, unsigned* __restrict__ mo, unsigned* __restrict__ mi) {
  int e = blockIdx.x * blockDim.x + threadIdx.x;
  if (e >= EHALF) return;
  int u = ei[e], v = ei[EFULL + e];
  #pragma unroll
  for (int h = 0; h < HH; ++h) {
    float lo = s1o[u*2+h] + s2o[v*2+h]; lo = lo > 0.f ? lo : ALPHA * lo;
    float li = s1i[v*2+h] + s2i[v*2+h]; li = li > 0.f ? li : ALPHA * li; // replicated bug: both halves use v
    att[(size_t)h*EFULL + e]         = lo;
    att[(size_t)h*EFULL + EHALF + e] = li;
    atomicMax(&mo[u*2+h], fkey(lo));
    atomicMax(&mi[v*2+h], fkey(li));
  }
}

// ---------------- exp + segment sum ----------------
__global__ void edge_exp_kernel(const int* __restrict__ ei,
                                float* __restrict__ att,
                                const unsigned* __restrict__ mo, const unsigned* __restrict__ mi,
                                float* __restrict__ deno, float* __restrict__ deni) {
  int e = blockIdx.x * blockDim.x + threadIdx.x;
  if (e >= EHALF) return;
  int u = ei[e], v = ei[EFULL + e];
  #pragma unroll
  for (int h = 0; h < HH; ++h) {
    float exo = expf(att[(size_t)h*EFULL + e]         - fdec(mo[u*2+h]));
    float exi = expf(att[(size_t)h*EFULL + EHALF + e] - fdec(mi[v*2+h]));
    att[(size_t)h*EFULL + e]         = exo;
    att[(size_t)h*EFULL + EHALF + e] = exi;
    atomicAdd(&deno[u*2+h], exo);
    atomicAdd(&deni[v*2+h], exi);
  }
}

// ---------------- normalize ----------------
__global__ void edge_norm_kernel(const int* __restrict__ ei, float* __restrict__ att,
                                 const float* __restrict__ deno, const float* __restrict__ deni) {
  int e = blockIdx.x * blockDim.x + threadIdx.x;
  if (e >= EHALF) return;
  int u = ei[e], v = ei[EFULL + e];
  #pragma unroll
  for (int h = 0; h < HH; ++h) {
    att[(size_t)h*EFULL + e]         /= deno[u*2+h];
    att[(size_t)h*EFULL + EHALF + e] /= deni[v*2+h];
  }
}

// ---------------- edge MLP: TDM weight DMA + WMMA bf16 + attention-weighted scatter ----------------
// grid.y in [0,4): f = y>>1 (0=out,1=in), h = y&1.  One wave32 per block; grid-stride tiles of 16 edges.
// Weights are pre-transposed: sw1 = [HID][FIN], sw2 = [DOUT][HID] (bf16).
__global__ __launch_bounds__(32)
void edge_mlp_kernel(const __bf16* __restrict__ xb, const float* __restrict__ edge_attr,
                     const int* __restrict__ ei,
                     const __bf16* __restrict__ w1o, const __bf16* __restrict__ w2o,
                     const __bf16* __restrict__ w1i, const __bf16* __restrict__ w2i,
                     const float* __restrict__ b1o, const float* __restrict__ b2o,
                     const float* __restrict__ b1i, const float* __restrict__ b2i,
                     const float* __restrict__ att, float* __restrict__ agg) {
  __shared__ __align__(16) __bf16 sw1[HID * FIN];   // 40KB, layer-1 weights (transposed)
  __shared__ __align__(16) __bf16 sw2[DOUT * HID];  // 16KB, layer-2 weights (transposed)
  __shared__ __align__(16) __bf16 sbuf[16 * FIN];   //  5KB, input tile; reused as hidden tile

  const int lane = threadIdx.x;
  const int f = blockIdx.y >> 1, h = blockIdx.y & 1;
  const __bf16* gw1 = (f == 0 ? w1o : w1i) + h * FIN * HID;
  const __bf16* gw2 = (f == 0 ? w2o : w2i) + h * HID * DOUT;
  const float*  b1  = (f == 0 ? b1o : b1i) + h * HID;
  const float*  b2  = (f == 0 ? b2o : b2i) + h * DOUT;

  // Tensor Data Mover: DMA both weight matrices into LDS, then wait on TENSORcnt.
  tdm_load_2B(lds_off(sw1), gw1, FIN * HID);
  tdm_load_2B(lds_off(sw2), gw2, HID * DOUT);
  __builtin_amdgcn_s_wait_tensorcnt(0);

  const int hi = lane >> 4, ln = lane & 15;
  const int base = (f == 0) ? HH * DOUT : 0;   // flow = concat([agg_in, agg_out])
  const int half = f ? EHALF : 0;
  const v8f vz = {0,0,0,0,0,0,0,0};
  __bf16* shid = sbuf;                         // alias: hidden tile after input consumed

  for (int t = blockIdx.x; t < EHALF / 16; t += gridDim.x) {
    // ---- stage input tile [16 x 160] (bf16) into LDS, coalesced, branch-free ----
    for (int row = 0; row < 16; ++row) {
      const int fl = half + t * 16 + row;
      const int src = ei[fl];
      // 128 bf16 of x row: 32 lanes x 8 bytes
      const uint2 xc = ((const uint2*)(xb + (size_t)src * DD))[lane];
      *(uint2*)&sbuf[row * FIN + lane * 4] = xc;
      // 32 f32 edge attrs: 1 per lane, converted in-flight
      sbuf[row * FIN + DD + lane] = (__bf16)edge_attr[(size_t)fl * EDIM + lane];
    }

    // ---- layer 1: [16x160] @ [160x128] -> [16x128] ----
    v8f acc[8];
    #pragma unroll
    for (int n = 0; n < 8; ++n) acc[n] = vz;
    for (int kt = 0; kt < 5; ++kt) {
      const v16bf av = fragA(sbuf + ln * FIN, kt * 32 + 8 * hi);
      #pragma unroll
      for (int n = 0; n < 8; ++n) {
        const v16bf bv = fragBT(sw1 + (n * 16 + ln) * FIN, kt * 32 + 8 * hi);
        acc[n] = wmma_bf16(av, bv, acc[n]);
      }
    }
    #pragma unroll
    for (int n = 0; n < 8; ++n) {
      int col = n * 16 + ln; float bb = b1[col];
      #pragma unroll
      for (int r = 0; r < 8; ++r) {
        float v = acc[n][r] + bb; v = v > 0.f ? v : 0.f;
        shid[(r + hi * 8) * HID + col] = (__bf16)v;
      }
    }

    // ---- layer 2: [16x128] @ [128x64] -> [16x64] ----
    v8f acc2[4];
    #pragma unroll
    for (int n = 0; n < 4; ++n) acc2[n] = vz;
    for (int kt = 0; kt < 4; ++kt) {
      const v16bf av = fragA(shid + ln * HID, kt * 32 + 8 * hi);
      #pragma unroll
      for (int n = 0; n < 4; ++n) {
        const v16bf bv = fragBT(sw2 + (n * 16 + ln) * HID, kt * 32 + 8 * hi);
        acc2[n] = wmma_bf16(av, bv, acc2[n]);
      }
    }

    // ---- attention-weighted segment-sum scatter ----
    float attv[8]; int nodev[8];
    #pragma unroll
    for (int r = 0; r < 8; ++r) {
      int e = t * 16 + r + hi * 8, fl = half + e;
      nodev[r] = ei[fl];
      attv[r]  = att[(size_t)h * EFULL + fl];
    }
    #pragma unroll
    for (int n = 0; n < 4; ++n) {
      int col = n * 16 + ln; float bb = b2[col];
      #pragma unroll
      for (int r = 0; r < 8; ++r)
        atomicAdd(&agg[(size_t)nodev[r] * (2 * HH * DOUT) + base + h * DOUT + col],
                  (acc2[n][r] + bb) * attv[r]);
    }
  }
}

// ---------------- node MLP: WMMA bf16, transposed weights streamed from L2-resident bf16 ----------------
// nw1 = [NHID][NHID] transposed, nw2 = [NOUT][NHID] transposed.
__global__ __launch_bounds__(32)
void node_mlp_kernel(const float* __restrict__ agg,
                     const __bf16* __restrict__ nw1, const __bf16* __restrict__ nw2,
                     const float* __restrict__ nb1, const float* __restrict__ nb2,
                     float* __restrict__ out) {
  __shared__ __align__(16) __bf16 sbuf[16 * NHID];  // 8KB: input tile, then hidden tile
  const int lane = threadIdx.x, hi = lane >> 4, ln = lane & 15;
  const v8f vz = {0,0,0,0,0,0,0,0};

  for (int t = blockIdx.x; t < NN / 16; t += gridDim.x) {
    // ---- stage [16x256] f32 agg tile to bf16 LDS (float4 coalesced) ----
    for (int row = 0; row < 16; ++row) {
      const float* ar = agg + (size_t)(t * 16 + row) * NHID;
      #pragma unroll
      for (int c = 0; c < 2; ++c) {
        float4 v4 = ((const float4*)ar)[lane + 32 * c];
        int o = row * NHID + (lane + 32 * c) * 4;
        sbuf[o + 0] = (__bf16)v4.x; sbuf[o + 1] = (__bf16)v4.y;
        sbuf[o + 2] = (__bf16)v4.z; sbuf[o + 3] = (__bf16)v4.w;
      }
    }

    // ---- layer 1: [16x256] @ [256x256] ----
    v8f acc[16];
    #pragma unroll
    for (int n = 0; n < 16; ++n) acc[n] = vz;
    for (int kt = 0; kt < 8; ++kt) {
      const v16bf av = fragA(sbuf + ln * NHID, kt * 32 + 8 * hi);
      #pragma unroll
      for (int n = 0; n < 16; ++n) {
        const v16bf bv = fragBT(nw1 + (size_t)(n * 16 + ln) * NHID, kt * 32 + 8 * hi);
        acc[n] = wmma_bf16(av, bv, acc[n]);
      }
    }
    #pragma unroll
    for (int n = 0; n < 16; ++n) {
      int col = n * 16 + ln; float bb = nb1[col];
      #pragma unroll
      for (int r = 0; r < 8; ++r) {
        float v = acc[n][r] + bb; v = v > 0.f ? v : 0.f;
        sbuf[(r + hi * 8) * NHID + col] = (__bf16)v;   // overwrite input tile with hidden
      }
    }

    // ---- layer 2: [16x256] @ [256x128] ----
    v8f acc2[8];
    #pragma unroll
    for (int n = 0; n < 8; ++n) acc2[n] = vz;
    for (int kt = 0; kt < 8; ++kt) {
      const v16bf av = fragA(sbuf + ln * NHID, kt * 32 + 8 * hi);
      #pragma unroll
      for (int n = 0; n < 8; ++n) {
        const v16bf bv = fragBT(nw2 + (size_t)(n * 16 + ln) * NHID, kt * 32 + 8 * hi);
        acc2[n] = wmma_bf16(av, bv, acc2[n]);
      }
    }
    #pragma unroll
    for (int n = 0; n < 8; ++n) {
      int col = n * 16 + ln; float bb = nb2[col];
      #pragma unroll
      for (int r = 0; r < 8; ++r)
        out[(size_t)(t * 16 + r + hi * 8) * NOUT + col] = acc2[n][r] + bb;
    }
  }
}

// ---------------- host launcher ----------------
extern "C" void kernel_launch(void* const* d_in, const int* in_sizes, int n_in,
                              void* d_out, int out_size, void* d_ws, size_t ws_size,
                              hipStream_t stream) {
  const float* x        = (const float*)d_in[0];
  const int*   ei       = (const int*)d_in[1];     // int32 (jax x64 disabled)
  const float* eattr    = (const float*)d_in[2];
  const float* w1_out   = (const float*)d_in[3];
  const float* b1_out   = (const float*)d_in[4];
  const float* w2_out   = (const float*)d_in[5];
  const float* b2_out   = (const float*)d_in[6];
  const float* w1_in    = (const float*)d_in[7];
  const float* b1_in    = (const float*)d_in[8];
  const float* w2_in    = (const float*)d_in[9];
  const float* b2_in    = (const float*)d_in[10];
  const float* aa_out   = (const float*)d_in[11];
  const float* aa_in    = (const float*)d_in[12];
  const float* node_w1  = (const float*)d_in[13];
  const float* node_b1  = (const float*)d_in[14];
  const float* node_w2  = (const float*)d_in[15];
  const float* node_b2  = (const float*)d_in[16];

  float* node_out = (float*)d_out;
  float* att      = (float*)d_out + (size_t)NN * NOUT;   // [H, E]

  // workspace carve (~132 MB total)
  char* p = (char*)d_ws;
  auto carve = [&](size_t bytes) { void* r = p; p += (bytes + 255) & ~(size_t)255; return r; };
  __bf16* xb   = (__bf16*)carve((size_t)NN * DD * 2);
  __bf16* w1ob = (__bf16*)carve((size_t)HH * FIN * HID * 2);   // transposed [H][HID][FIN]
  __bf16* w2ob = (__bf16*)carve((size_t)HH * HID * DOUT * 2);  // transposed [H][DOUT][HID]
  __bf16* w1ib = (__bf16*)carve((size_t)HH * FIN * HID * 2);
  __bf16* w2ib = (__bf16*)carve((size_t)HH * HID * DOUT * 2);
  __bf16* nw1b = (__bf16*)carve((size_t)NHID * NHID * 2);      // transposed [NHID][NHID]
  __bf16* nw2b = (__bf16*)carve((size_t)NHID * NOUT * 2);      // transposed [NOUT][NHID]
  float* s1o   = (float*)carve((size_t)NN * HH * 4);
  float* s2o   = (float*)carve((size_t)NN * HH * 4);
  float* s1i   = (float*)carve((size_t)NN * HH * 4);
  float* s2i   = (float*)carve((size_t)NN * HH * 4);
  unsigned* mo = (unsigned*)carve((size_t)NN * HH * 4);
  unsigned* mi = (unsigned*)carve((size_t)NN * HH * 4);
  float* deno  = (float*)carve((size_t)NN * HH * 4);
  float* deni  = (float*)carve((size_t)NN * HH * 4);
  float* agg   = (float*)carve((size_t)NN * 2 * HH * DOUT * 4);  // [N,256]

  cvt_bf16_kernel<<<(NN * DD + 255) / 256, 256, 0, stream>>>(x, xb, NN * DD);
  auto cvtT = [&](const float* in, __bf16* out, int R, int C, int nm) {
    cvt_bf16_T_kernel<<<(nm * R * C + 255) / 256, 256, 0, stream>>>(in, out, R, C, nm);
  };
  cvtT(w1_out, w1ob, FIN, HID, HH);
  cvtT(w2_out, w2ob, HID, DOUT, HH);
  cvtT(w1_in,  w1ib, FIN, HID, HH);
  cvtT(w2_in,  w2ib, HID, DOUT, HH);
  cvtT(node_w1, nw1b, NHID, NHID, 1);
  cvtT(node_w2, nw2b, NHID, NOUT, 1);

  init_soft_kernel<<<(NN * HH + 255) / 256, 256, 0, stream>>>(mo, mi, deno, deni);
  (void)hipMemsetAsync(agg, 0, (size_t)NN * 2 * HH * DOUT * 4, stream);

  node_scores_kernel<<<(NN + 255) / 256, 256, 0, stream>>>(x, aa_out, aa_in, s1o, s2o, s1i, s2i);
  edge_logits_kernel<<<(EHALF + 255) / 256, 256, 0, stream>>>(ei, s1o, s2o, s1i, s2i, att, mo, mi);
  edge_exp_kernel<<<(EHALF + 255) / 256, 256, 0, stream>>>(ei, att, mo, mi, deno, deni);
  edge_norm_kernel<<<(EHALF + 255) / 256, 256, 0, stream>>>(ei, att, deno, deni);

  edge_mlp_kernel<<<dim3(1024, 4), 32, 0, stream>>>(xb, eattr, ei, w1ob, w2ob, w1ib, w2ib,
                                                    b1_out, b2_out, b1_in, b2_in, att, agg);
  node_mlp_kernel<<<2048, 32, 0, stream>>>(agg, nw1b, nw2b, node_b1, node_b2, node_out);
}
